// TransformerBlockTF_16028817949065
// MI455X (gfx1250) — matrix operand order repaired and checked by reading, not compiled, not run on hardware
//
#include <hip/hip_runtime.h>
#include <hip/hip_bf16.h>
#include <math.h>

#define D_MODEL 768
#define N_HEADS 12
#define HEAD_DIM 64
#define D_FF 3072
#define SEQ 1024
#define BATCH 4
#define NTOK (BATCH * SEQ)   // 4096

typedef __bf16 bf16;
typedef __attribute__((ext_vector_type(16))) __bf16 v16bf;
typedef __attribute__((ext_vector_type(8)))  __bf16 v8bf;
typedef __attribute__((ext_vector_type(8)))  float  v8f;

// ---------------------------------------------------------------------------
// WMMA fragment helpers (CDNA5 16x16x32 bf16, wave32)
//
// A (16x32, MxK): lane (m = lane&15, hi = lane>>4).
//   elems 0..7  -> K = k0 + hi*8 + j
//   elems 8..15 -> K = k0 + 16 + hi*8 + (j-8)
// B (32x16, KxN) from BT stored [N][K] row-major (stride ld):
//   lane n = lane&15; elems j -> same K mapping as A (k0 + hi*8 + j, k0 + 16 + hi*8 + (j-8))
// C/D (16x16 f32): elem r -> (M = hi*8 + r, N = lane&15)
// ---------------------------------------------------------------------------
__device__ __forceinline__ v16bf make_frag(v8bf lo, v8bf hi) {
  return __builtin_shufflevector(lo, hi, 0,1,2,3,4,5,6,7,8,9,10,11,12,13,14,15);
}

__device__ __forceinline__ v16bf load_a(const bf16* A, int ld, int row, int k0, int hi) {
  const bf16* p = A + (size_t)row * ld + k0 + hi * 8;
  v8bf lo = *(const v8bf*)p;
  v8bf h  = *(const v8bf*)(p + 16);
  return make_frag(lo, h);
}

__device__ __forceinline__ v16bf load_b(const bf16* BT, int ld, int col, int k0, int hi) {
  const bf16* p = BT + (size_t)col * ld + k0 + hi * 8;
  v8bf lo = *(const v8bf*)p;
  v8bf h  = *(const v8bf*)(p + 16);
  return make_frag(lo, h);
}

__device__ __forceinline__ v8f wmma_bf16(v16bf a, v16bf b, v8f c) {
  return __builtin_amdgcn_wmma_f32_16x16x32_bf16(false, a, false, b, (short)0, c, false, false);
}

// ---------------------------------------------------------------------------
// LayerNorm (f32 in) -> bf16 out.  One block per token, 256 threads x 3 elems.
// ---------------------------------------------------------------------------
__global__ __launch_bounds__(256) void ln_kernel(const float* __restrict__ x,
                                                 const float* __restrict__ g,
                                                 const float* __restrict__ b,
                                                 bf16* __restrict__ out) {
  int tok = blockIdx.x;
  int t = threadIdx.x;
  const float* xr = x + (size_t)tok * D_MODEL;
  float v0 = xr[t], v1 = xr[t + 256], v2 = xr[t + 512];
  __shared__ float ss[256], sq[256];
  ss[t] = v0 + v1 + v2;
  sq[t] = v0 * v0 + v1 * v1 + v2 * v2;
  __syncthreads();
  for (int o = 128; o > 0; o >>= 1) {
    if (t < o) { ss[t] += ss[t + o]; sq[t] += sq[t + o]; }
    __syncthreads();
  }
  float mu   = ss[0] * (1.0f / D_MODEL);
  float var  = sq[0] * (1.0f / D_MODEL) - mu * mu;
  float rstd = rsqrtf(var + 1e-5f);
  bf16* orow = out + (size_t)tok * D_MODEL;
  orow[t]       = (bf16)((v0 - mu) * rstd * g[t]       + b[t]);
  orow[t + 256] = (bf16)((v1 - mu) * rstd * g[t + 256] + b[t + 256]);
  orow[t + 512] = (bf16)((v2 - mu) * rstd * g[t + 512] + b[t + 512]);
}

// ---------------------------------------------------------------------------
// Weight transpose + f32->bf16: w[K][N] -> wT[N][K]
// ---------------------------------------------------------------------------
__global__ __launch_bounds__(256) void transpose_kernel(const float* __restrict__ w,
                                                        bf16* __restrict__ wT,
                                                        int K, int N) {
  size_t i = (size_t)blockIdx.x * 256 + threadIdx.x;
  if (i < (size_t)K * N) {
    int n = (int)(i / K);
    int k = (int)(i % K);
    wT[i] = (bf16)w[(size_t)k * N + n];
  }
}

// ---------------------------------------------------------------------------
// WMMA GEMM: C[M,N] = A[M,K](bf16) * BT[N,K](bf16)^T + bias
// wave -> 32x64 output tile (2 M-frags x 4 N-frags), register double-buffered
// k-loop. Block 256 thr = 8 waves -> 256 rows.
// EPI==0: outf = acc + bias + res (f32)
// EPI==1: outb = gelu(acc+bias) (bf16)
// EPI==2: QKV scatter -> q[B,H,S,64], k[B,H,S,64], vt[B,H,64,S] (bf16)
// ---------------------------------------------------------------------------
template <int EPI>
__global__ __launch_bounds__(256) void gemm_wmma(const bf16* __restrict__ A,
                                                 const bf16* __restrict__ BT,
                                                 int N, int K,
                                                 const float* __restrict__ bias,
                                                 const float* __restrict__ res,
                                                 float* __restrict__ outf,
                                                 bf16* __restrict__ outb,
                                                 bf16* __restrict__ q,
                                                 bf16* __restrict__ kkp,
                                                 bf16* __restrict__ vt) {
  int wave = threadIdx.x >> 5, lane = threadIdx.x & 31;
  int mlane = lane & 15, hi = lane >> 4;
  int row0 = blockIdx.y * 256 + wave * 32;
  int col0 = blockIdx.x * 64;
  int arow0 = row0 + mlane;
  int arow1 = row0 + 16 + mlane;

  v8f acc[2][4] = {};

  // ---- prologue: fragments for k0 = 0 ----
  v16bf a0 = load_a(A, K, arow0, 0, hi);
  v16bf a1 = load_a(A, K, arow1, 0, hi);
  v16bf b0 = load_b(BT, K, col0 +  0 + mlane, 0, hi);
  v16bf b1 = load_b(BT, K, col0 + 16 + mlane, 0, hi);
  v16bf b2 = load_b(BT, K, col0 + 32 + mlane, 0, hi);
  v16bf b3 = load_b(BT, K, col0 + 48 + mlane, 0, hi);

  // ---- pipelined main loop: prefetch k0, compute k0-32 ----
  for (int k0 = 32; k0 < K; k0 += 32) {
    v16bf na0 = load_a(A, K, arow0, k0, hi);
    v16bf na1 = load_a(A, K, arow1, k0, hi);
    v16bf nb0 = load_b(BT, K, col0 +  0 + mlane, k0, hi);
    v16bf nb1 = load_b(BT, K, col0 + 16 + mlane, k0, hi);
    v16bf nb2 = load_b(BT, K, col0 + 32 + mlane, k0, hi);
    v16bf nb3 = load_b(BT, K, col0 + 48 + mlane, k0, hi);

    acc[0][0] = wmma_bf16(a0, b0, acc[0][0]);
    acc[1][0] = wmma_bf16(a1, b0, acc[1][0]);
    acc[0][1] = wmma_bf16(a0, b1, acc[0][1]);
    acc[1][1] = wmma_bf16(a1, b1, acc[1][1]);
    acc[0][2] = wmma_bf16(a0, b2, acc[0][2]);
    acc[1][2] = wmma_bf16(a1, b2, acc[1][2]);
    acc[0][3] = wmma_bf16(a0, b3, acc[0][3]);
    acc[1][3] = wmma_bf16(a1, b3, acc[1][3]);

    a0 = na0; a1 = na1; b0 = nb0; b1 = nb1; b2 = nb2; b3 = nb3;
  }
  // ---- drain last k-step ----
  acc[0][0] = wmma_bf16(a0, b0, acc[0][0]);
  acc[1][0] = wmma_bf16(a1, b0, acc[1][0]);
  acc[0][1] = wmma_bf16(a0, b1, acc[0][1]);
  acc[1][1] = wmma_bf16(a1, b1, acc[1][1]);
  acc[0][2] = wmma_bf16(a0, b2, acc[0][2]);
  acc[1][2] = wmma_bf16(a1, b2, acc[1][2]);
  acc[0][3] = wmma_bf16(a0, b3, acc[0][3]);
  acc[1][3] = wmma_bf16(a1, b3, acc[1][3]);

  // ---- epilogue ----
#pragma unroll
  for (int gset = 0; gset < 2; ++gset) {
    int rbase = row0 + gset * 16 + hi * 8;
#pragma unroll
    for (int t = 0; t < 4; ++t) {
      int col = col0 + t * 16 + mlane;
      float bv = bias[col];
      if constexpr (EPI == 2) {
        int part = col / D_MODEL;
        int c    = col - part * D_MODEL;
        int h    = c >> 6;
        int d    = c & 63;
#pragma unroll
        for (int r = 0; r < 8; ++r) {
          int row = rbase + r;
          float v = acc[gset][t][r] + bv;
          int bb = row >> 10, s = row & (SEQ - 1);
          size_t bh = (size_t)bb * N_HEADS + h;
          if (part == 0)      q[(bh * SEQ + s) * HEAD_DIM + d]   = (bf16)v;
          else if (part == 1) kkp[(bh * SEQ + s) * HEAD_DIM + d] = (bf16)v;
          else                vt[(bh * HEAD_DIM + d) * SEQ + s]  = (bf16)v;
        }
      } else {
#pragma unroll
        for (int r = 0; r < 8; ++r) {
          float v = acc[gset][t][r] + bv;
          size_t idx = (size_t)(rbase + r) * N + col;
          if constexpr (EPI == 0) {
            outf[idx] = v + res[idx];
          } else {
            float cc = 0.7978845608028654f * (v + 0.044715f * v * v * v);
            outb[idx] = (bf16)(0.5f * v * (1.0f + tanhf(cc)));
          }
        }
      }
    }
  }
}

// ---------------------------------------------------------------------------
// Flash attention, causal. 1 wave (32 thr) per 16-row query block.
// grid.x = B*H*(S/16). Key blocks of 32; WMMA for QK^T and P*V.
// V B-fragments are prefetched before the softmax VALU work so the loads
// overlap exp/shuffle compute.
// Output Y as bf16 [NTOK, D_MODEL] (row = b*S+s, col = h*64+d).
// ---------------------------------------------------------------------------
__global__ __launch_bounds__(32) void attn_kernel(const bf16* __restrict__ Qg,
                                                  const bf16* __restrict__ Kg,
                                                  const bf16* __restrict__ VTg,
                                                  bf16* __restrict__ Y) {
  __shared__ bf16 sP[16 * 32];
  int lane = threadIdx.x;
  int mlane = lane & 15, hi = lane >> 4;
  int qblk = blockIdx.x & 63;
  int bh   = blockIdx.x >> 6;           // 0..47
  const bf16* Q  = Qg  + (size_t)bh * SEQ * HEAD_DIM;
  const bf16* Kp = Kg  + (size_t)bh * SEQ * HEAD_DIM;
  const bf16* VT = VTg + (size_t)bh * HEAD_DIM * SEQ;
  int qrow0 = qblk * 16;

  v16bf aq0 = load_a(Q, HEAD_DIM, qrow0 + mlane, 0, hi);
  v16bf aq1 = load_a(Q, HEAD_DIM, qrow0 + mlane, 32, hi);

  v8f o0 = {}, o1 = {}, o2 = {}, o3 = {};
  float mrow[8], lrow[8];
#pragma unroll
  for (int r = 0; r < 8; ++r) { mrow[r] = -3.0e38f; lrow[r] = 0.0f; }

  int kend = qrow0 + 16;                 // keys 0 .. qrow0+15 can be live
  for (int kb = 0; kb < kend; kb += 32) {
    // ---- scores: S = (Q K^T) for key cols [kb, kb+32) ----
    v8f s0 = {}, s1 = {};
    s0 = wmma_bf16(aq0, load_b(Kp, HEAD_DIM, kb + mlane, 0, hi), s0);
    s0 = wmma_bf16(aq1, load_b(Kp, HEAD_DIM, kb + mlane, 32, hi), s0);
    s1 = wmma_bf16(aq0, load_b(Kp, HEAD_DIM, kb + 16 + mlane, 0, hi), s1);
    s1 = wmma_bf16(aq1, load_b(Kp, HEAD_DIM, kb + 16 + mlane, 32, hi), s1);

    // ---- prefetch V chunk B-fragments (independent of softmax) ----
    v16bf vb0 = load_b(VT, SEQ, 0  + mlane, kb, hi);
    v16bf vb1 = load_b(VT, SEQ, 16 + mlane, kb, hi);
    v16bf vb2 = load_b(VT, SEQ, 32 + mlane, kb, hi);
    v16bf vb3 = load_b(VT, SEQ, 48 + mlane, kb, hi);

    // ---- online softmax ----
    float p0[8], p1[8];
#pragma unroll
    for (int r = 0; r < 8; ++r) {
      int qrow = qrow0 + hi * 8 + r;
      float x0 = (kb + mlane      <= qrow) ? s0[r] * 0.125f : -1.0e30f;
      float x1 = (kb + 16 + mlane <= qrow) ? s1[r] * 0.125f : -1.0e30f;
      float mx = fmaxf(x0, x1);
#pragma unroll
      for (int off = 1; off < 16; off <<= 1) mx = fmaxf(mx, __shfl_xor(mx, off, 32));
      float mnew = fmaxf(mrow[r], mx);
      float sc = __expf(mrow[r] - mnew);
      mrow[r] = mnew;
      float e0 = __expf(x0 - mnew), e1 = __expf(x1 - mnew);
      float sum = e0 + e1;
#pragma unroll
      for (int off = 1; off < 16; off <<= 1) sum += __shfl_xor(sum, off, 32);
      lrow[r] = lrow[r] * sc + sum;
      o0[r] *= sc; o1[r] *= sc; o2[r] *= sc; o3[r] *= sc;
      p0[r] = e0; p1[r] = e1;
    }

    // ---- C-layout P -> A-layout fragment via LDS ----
#pragma unroll
    for (int r = 0; r < 8; ++r) {
      sP[(hi * 8 + r) * 32 + mlane]      = (bf16)p0[r];
      sP[(hi * 8 + r) * 32 + 16 + mlane] = (bf16)p1[r];
    }
    __syncthreads();
    v16bf pa = load_a(sP, 32, mlane, 0, hi);
    __syncthreads();

    // ---- O += P * V ----
    o0 = wmma_bf16(pa, vb0, o0);
    o1 = wmma_bf16(pa, vb1, o1);
    o2 = wmma_bf16(pa, vb2, o2);
    o3 = wmma_bf16(pa, vb3, o3);
  }

  int bb = bh / N_HEADS, h = bh % N_HEADS;
#pragma unroll
  for (int r = 0; r < 8; ++r) {
    int s = qrow0 + hi * 8 + r;
    float inv = 1.0f / lrow[r];
    size_t base = ((size_t)(bb * SEQ + s)) * D_MODEL + h * HEAD_DIM + mlane;
    Y[base + 0]  = (bf16)(o0[r] * inv);
    Y[base + 16] = (bf16)(o1[r] * inv);
    Y[base + 32] = (bf16)(o2[r] * inv);
    Y[base + 48] = (bf16)(o3[r] * inv);
  }
}

// ---------------------------------------------------------------------------
extern "C" void kernel_launch(void* const* d_in, const int* in_sizes, int n_in,
                              void* d_out, int out_size, void* d_ws, size_t ws_size,
                              hipStream_t stream) {
  const float* x      = (const float*)d_in[0];
  const float* ln1_g  = (const float*)d_in[1];
  const float* ln1_b  = (const float*)d_in[2];
  const float* ln2_g  = (const float*)d_in[3];
  const float* ln2_b  = (const float*)d_in[4];
  const float* w_attn = (const float*)d_in[5];
  const float* b_attn = (const float*)d_in[6];
  const float* w_proj = (const float*)d_in[7];
  const float* b_proj = (const float*)d_in[8];
  const float* w_fc   = (const float*)d_in[9];
  const float* b_fc   = (const float*)d_in[10];
  const float* w_mlp  = (const float*)d_in[11];
  const float* b_mlp  = (const float*)d_in[12];

  char* ws = (char*)d_ws;
  size_t off = 0;
  auto alloc = [&](size_t bytes) -> void* {
    void* p = ws + off;
    off += (bytes + 255) & ~(size_t)255;
    return p;
  };
  bf16* xn1    = (bf16*)alloc((size_t)NTOK * D_MODEL * 2);
  bf16* wqkvT  = (bf16*)alloc((size_t)3 * D_MODEL * D_MODEL * 2);
  bf16* qb     = (bf16*)alloc((size_t)NTOK * D_MODEL * 2);
  bf16* kb     = (bf16*)alloc((size_t)NTOK * D_MODEL * 2);
  bf16* vtb    = (bf16*)alloc((size_t)NTOK * D_MODEL * 2);
  bf16* yb     = (bf16*)alloc((size_t)NTOK * D_MODEL * 2);
  bf16* wprojT = (bf16*)alloc((size_t)D_MODEL * D_MODEL * 2);
  float* res1  = (float*)alloc((size_t)NTOK * D_MODEL * 4);
  bf16* xn2    = (bf16*)alloc((size_t)NTOK * D_MODEL * 2);
  bf16* wfcT   = (bf16*)alloc((size_t)D_FF * D_MODEL * 2);
  bf16* wmlpT  = (bf16*)alloc((size_t)D_MODEL * D_FF * 2);
  bf16* hb     = qb;  // alias: q,k,vt,y (4*6.29MB) dead before FF act (25.2MB)

  // 1) LN1 -> bf16
  ln_kernel<<<NTOK, 256, 0, stream>>>(x, ln1_g, ln1_b, xn1);
  // 2) weight transposes + bf16 convert
  transpose_kernel<<<(3 * D_MODEL * D_MODEL) / 256, 256, 0, stream>>>(w_attn, wqkvT, D_MODEL, 3 * D_MODEL);
  transpose_kernel<<<(D_MODEL * D_MODEL) / 256, 256, 0, stream>>>(w_proj, wprojT, D_MODEL, D_MODEL);
  transpose_kernel<<<(D_MODEL * D_FF) / 256, 256, 0, stream>>>(w_fc, wfcT, D_MODEL, D_FF);
  transpose_kernel<<<(D_FF * D_MODEL) / 256, 256, 0, stream>>>(w_mlp, wmlpT, D_FF, D_MODEL);
  // 3) QKV projection (scatter epilogue)
  gemm_wmma<2><<<dim3((3 * D_MODEL) / 64, NTOK / 256), 256, 0, stream>>>(
      xn1, wqkvT, 3 * D_MODEL, D_MODEL, b_attn,
      nullptr, nullptr, nullptr, qb, kb, vtb);
  // 4) causal flash attention
  attn_kernel<<<BATCH * N_HEADS * (SEQ / 16), 32, 0, stream>>>(qb, kb, vtb, yb);
  // 5) output projection + residual -> res1 (f32)
  gemm_wmma<0><<<dim3(D_MODEL / 64, NTOK / 256), 256, 0, stream>>>(
      yb, wprojT, D_MODEL, D_MODEL, b_proj, x, res1, nullptr,
      nullptr, nullptr, nullptr);
  // 6) LN2 -> bf16
  ln_kernel<<<NTOK, 256, 0, stream>>>(res1, ln2_g, ln2_b, xn2);
  // 7) FC + gelu -> bf16
  gemm_wmma<1><<<dim3(D_FF / 64, NTOK / 256), 256, 0, stream>>>(
      xn2, wfcT, D_FF, D_MODEL, b_fc, nullptr, nullptr, hb,
      nullptr, nullptr, nullptr);
  // 8) MLP proj + residual -> d_out (f32)
  gemm_wmma<0><<<dim3(D_MODEL / 64, NTOK / 256), 256, 0, stream>>>(
      hb, wmlpT, D_MODEL, D_FF, b_mlp, res1, (float*)d_out, nullptr,
      nullptr, nullptr, nullptr);
}
